// TernaryDenseLayer_27522150433243
// MI455X (gfx1250) — matrix operand order repaired and checked by reading, not compiled, not run on hardware
//
#include <hip/hip_runtime.h>
#include <hip/hip_bf16.h>

typedef __attribute__((ext_vector_type(16))) __bf16 v16bf;
typedef __attribute__((ext_vector_type(8)))  __bf16 v8bf;
typedef __attribute__((ext_vector_type(4)))  __bf16 v4bf;
typedef __attribute__((ext_vector_type(8)))  float  v8f;

// ---------------- reductions (deterministic two-pass) ----------------

__device__ __forceinline__ float waveReduceSum(float v) {
#pragma unroll
  for (int off = 16; off > 0; off >>= 1) v += __shfl_down(v, off, 32);
  return v;
}

__device__ __forceinline__ float blockReduceSum(float v) {
  __shared__ float smem[8];
  __syncthreads();                       // safe reuse across multiple calls
  const int lane = threadIdx.x & 31;
  const int wave = threadIdx.x >> 5;
  v = waveReduceSum(v);
  if (lane == 0) smem[wave] = v;
  __syncthreads();
  if (wave == 0) {
    v = (lane < (int)(blockDim.x >> 5)) ? smem[lane] : 0.0f;
    v = waveReduceSum(v);
  }
  return v;                              // valid in wave 0, lane 0
}

__global__ void k_abs_partial(const float4* __restrict__ W4,
                              float* __restrict__ part, int n4) {
  float s = 0.0f;
  for (int i = blockIdx.x * blockDim.x + threadIdx.x; i < n4;
       i += gridDim.x * blockDim.x) {
    float4 v = W4[i];
    s += fabsf(v.x) + fabsf(v.y) + fabsf(v.z) + fabsf(v.w);
  }
  s = blockReduceSum(s);
  if (threadIdx.x == 0) part[blockIdx.x] = s;
}

__global__ void k_reduce_abs(const float* __restrict__ part,
                             float* __restrict__ sc, int nparts) {
  float s = 0.0f;
  for (int i = threadIdx.x; i < nparts; i += blockDim.x) s += part[i];
  s = blockReduceSum(s);
  if (threadIdx.x == 0) sc[0] = s;       // sum(|W|)
}

__global__ void k_alpha_partial(const float4* __restrict__ W4,
                                const float* __restrict__ sc,
                                float* __restrict__ pA, float* __restrict__ pC,
                                int n4, float invsize) {
  const float thr = 0.7f * sc[0] * invsize;
  float sa = 0.0f, cnt = 0.0f;
  for (int i = blockIdx.x * blockDim.x + threadIdx.x; i < n4;
       i += gridDim.x * blockDim.x) {
    float4 v = W4[i];
    float w;
    w = fabsf(v.x); if (w > thr) { sa += w; cnt += 1.0f; }
    w = fabsf(v.y); if (w > thr) { sa += w; cnt += 1.0f; }
    w = fabsf(v.z); if (w > thr) { sa += w; cnt += 1.0f; }
    w = fabsf(v.w); if (w > thr) { sa += w; cnt += 1.0f; }
  }
  sa  = blockReduceSum(sa);
  cnt = blockReduceSum(cnt);
  if (threadIdx.x == 0) { pA[blockIdx.x] = sa; pC[blockIdx.x] = cnt; }
}

__global__ void k_alpha_reduce(const float* __restrict__ pA,
                               const float* __restrict__ pC,
                               float* __restrict__ sc, int nparts) {
  float sa = 0.0f, cc = 0.0f;
  for (int i = threadIdx.x; i < nparts; i += blockDim.x) {
    sa += pA[i]; cc += pC[i];
  }
  sa = blockReduceSum(sa);
  cc = blockReduceSum(cc);
  if (threadIdx.x == 0) { sc[1] = sa; sc[2] = cc; }   // alpha = sc[1]/sc[2]
}

// ---------------- data preparation ----------------

// x (f32) -> bf16, 4 elements/thread
__global__ void k_cvt_x(const float4* __restrict__ X4,
                        v4bf* __restrict__ xb4, int n4) {
  int i = blockIdx.x * blockDim.x + threadIdx.x;
  if (i < n4) {
    float4 v = X4[i];
    v4bf o;
    o[0] = (__bf16)v.x; o[1] = (__bf16)v.y;
    o[2] = (__bf16)v.z; o[3] = (__bf16)v.w;
    xb4[i] = o;
  }
}

// ternarize W and transpose: Wt[n][k] = tern(W[k][n]) in bf16 (exact)
__global__ void k_tern_tr(const float* __restrict__ W,
                          const float* __restrict__ sc,
                          __bf16* __restrict__ Wt, int K, int N,
                          float invsize) {
  __shared__ float tile[32][33];
  const float thr = 0.7f * sc[0] * invsize;
  const int n0 = blockIdx.x * 32;
  const int k0 = blockIdx.y * 32;
  const int tx = threadIdx.x;      // 0..31
  const int ty = threadIdx.y;      // 0..7
#pragma unroll
  for (int r = 0; r < 4; ++r) {
    int kl = ty + 8 * r;
    tile[kl][tx] = W[(size_t)(k0 + kl) * N + (n0 + tx)];
  }
  __syncthreads();
#pragma unroll
  for (int r = 0; r < 4; ++r) {
    int nl = ty + 8 * r;
    float w = tile[tx][nl];                     // W[k0+tx][n0+nl]
    float t = (w >= thr) ? 1.0f : ((w <= -thr) ? -1.0f : 0.0f);
    Wt[(size_t)(n0 + nl) * K + (k0 + tx)] = (__bf16)t;
  }
}

// ---------------- WMMA GEMM: out = alpha * (x @ tern) + b ----------------
// Wave tile 64x64 (4x4 of 16x16). Block = 256 thr = 8 waves -> 128x256 tile.

#define TM 4
#define TN 4

__global__ __launch_bounds__(256) void k_gemm(
    const __bf16* __restrict__ X, const __bf16* __restrict__ Wt,
    const float* __restrict__ bias, const float* __restrict__ sc,
    float* __restrict__ out, int M, int N, int K) {
  const int lane  = threadIdx.x & 31;
  const int wave  = threadIdx.x >> 5;
  const int wm    = wave >> 2;                 // 0..1
  const int wn    = wave & 3;                  // 0..3
  const int gm    = blockIdx.y * 128 + wm * 64;
  const int gn    = blockIdx.x * 256 + wn * 64;
  const int mrow  = lane & 15;
  const int half  = lane >> 4;
  const int akoff = half * 16;                 // A: 16-bit 16x32 layout
  const int bkoff = half * 16;                 // B: 16-bit 32x16 layout
  const float alpha = sc[1] / sc[2];

  const __bf16* Abase[TM];
  const __bf16* Bbase[TN];
#pragma unroll
  for (int i = 0; i < TM; ++i)
    Abase[i] = X + (size_t)(gm + 16 * i + mrow) * K;
#pragma unroll
  for (int j = 0; j < TN; ++j)
    Bbase[j] = Wt + (size_t)(gn + 16 * j + mrow) * K;

  v8f c[TM][TN] = {};

  for (int k0 = 0; k0 < K; k0 += 32) {
    v16bf a[TM], bfr[TN];
#pragma unroll
    for (int i = 0; i < TM; ++i) {
      v8bf lo = *(const v8bf*)(Abase[i] + k0 + akoff);
      v8bf hi = *(const v8bf*)(Abase[i] + k0 + 8 + akoff);
      a[i] = __builtin_shufflevector(lo, hi, 0, 1, 2, 3, 4, 5, 6, 7,
                                     8, 9, 10, 11, 12, 13, 14, 15);
    }
#pragma unroll
    for (int j = 0; j < TN; ++j)
      bfr[j] = *(const v16bf*)(Bbase[j] + k0 + bkoff);
#pragma unroll
    for (int i = 0; i < TM; ++i)
#pragma unroll
      for (int j = 0; j < TN; ++j)
        c[i][j] = __builtin_amdgcn_wmma_f32_16x16x32_bf16(
            false, a[i], false, bfr[j], (short)0, c[i][j], false, false);
  }

  // epilogue: C/D layout -> lane n = lane&15, rows M = r + 8*half
#pragma unroll
  for (int j = 0; j < TN; ++j) {
    const int col = gn + 16 * j + mrow;
    const float bb = bias[col];
#pragma unroll
    for (int i = 0; i < TM; ++i) {
#pragma unroll
      for (int r = 0; r < 8; ++r) {
        const int row = gm + 16 * i + r + 8 * half;
        out[(size_t)row * N + col] = alpha * c[i][j][r] + bb;
      }
    }
  }
}

// ---------------- host launcher ----------------

extern "C" void kernel_launch(void* const* d_in, const int* in_sizes, int n_in,
                              void* d_out, int out_size, void* d_ws, size_t ws_size,
                              hipStream_t stream) {
  const float* x = (const float*)d_in[0];
  const float* W = (const float*)d_in[1];
  const float* b = (const float*)d_in[2];
  float* out = (float*)d_out;

  const int N = in_sizes[2];             // 4096
  const int K = in_sizes[1] / N;         // 4096
  const int M = in_sizes[0] / K;         // 8192
  const size_t nW = (size_t)K * N;
  const size_t nX = (size_t)M * K;
  const float invW = 1.0f / (float)nW;

  char*  ws = (char*)d_ws;
  float* sc = (float*)ws;                      // [0]=sum|W| [1]=masked sum [2]=count
  float* p0 = (float*)(ws + 1024);             // 1024 partials
  float* p1 = p0 + 1024;
  float* p2 = p1 + 1024;
  __bf16* xb = (__bf16*)(ws + (64 << 10));     // 64 KB offset, 64 MB
  __bf16* wt = xb + nX;                        // 32 MB

  const int RB = 1024, RT = 256;
  k_abs_partial <<<RB, RT, 0, stream>>>((const float4*)W, p0, (int)(nW >> 2));
  k_reduce_abs  <<<1,  RT, 0, stream>>>(p0, sc, RB);
  k_alpha_partial<<<RB, RT, 0, stream>>>((const float4*)W, sc, p1, p2,
                                         (int)(nW >> 2), invW);
  k_alpha_reduce<<<1,  RT, 0, stream>>>(p1, p2, sc, RB);

  k_cvt_x<<<(int)((nX >> 2) / 256), 256, 0, stream>>>((const float4*)x,
                                                      (v4bf*)xb, (int)(nX >> 2));
  dim3 tb(32, 8);
  k_tern_tr<<<dim3(N / 32, K / 32), tb, 0, stream>>>(W, sc, wt, K, N, invW);

  dim3 gg(N / 256, M / 128);
  k_gemm<<<gg, 256, 0, stream>>>(xb, wt, b, sc, out, M, N, K);
}
